// FlashAttention_3427383902670
// MI455X (gfx1250) — matrix operand order repaired and checked by reading, not compiled, not run on hardware
//
#include <hip/hip_runtime.h>
#include <hip/hip_bf16.h>

typedef __attribute__((ext_vector_type(16))) __bf16 v16bf;
typedef __attribute__((ext_vector_type(8)))  __bf16 v8bf;
typedef __attribute__((ext_vector_type(4)))  __bf16 v4bf;
typedef __attribute__((ext_vector_type(8)))  float  v8f;

#define T_TOK 2176
#define TPAD  2208
#define HSZ   2048
#define NOUT  3072
#define NH    16
#define NKV   4
#define HD    128
#define MMETA 128
#define WWIN  512
#define QK_SCALE 0.08838834764831845f   // 128^-0.5
#define MAXC  656                       // max key columns per query tile
#define PCOLS 672                       // MAXC padded to multiple of 32
#define LDSW  40                        // padded LDS row width (bf16): 80B, 16B aligned

// ---------------- CDNA5 async copy helpers (ASYNCcnt-tracked, bypass VGPRs) ---
__device__ __forceinline__ void async_b128(uint32_t lds_off, const void* gptr)
{
    asm volatile("global_load_async_to_lds_b128 %0, %1, off"
                 :: "v"(lds_off), "v"((unsigned long long)(uintptr_t)gptr)
                 : "memory");
}
__device__ __forceinline__ void wait_async0()
{
    asm volatile("s_wait_asynccnt 0" ::: "memory");
}

// ---------------------------------------------------------------- fp32 -> bf16
__global__ void k_cvt_hidden(const float* __restrict__ x, __bf16* __restrict__ y, int n4)
{
    int i = blockIdx.x * blockDim.x + threadIdx.x;
    if (i >= n4) return;
    float4 v = ((const float4*)x)[i];
    v4bf o = { (__bf16)v.x, (__bf16)v.y, (__bf16)v.z, (__bf16)v.w };
    ((v4bf*)y)[i] = o;
}

// ------------------------------------------- W (K x N) -> Wt bf16 (N x K), LDS tile
__global__ void k_transpose_w(const float* __restrict__ w, __bf16* __restrict__ wt)
{
    __shared__ float tile[32][33];
    const int n0 = blockIdx.x * 32, k0 = blockIdx.y * 32;
    const int tx = threadIdx.x, ty = threadIdx.y;
    tile[ty][tx] = w[(size_t)(k0 + ty) * NOUT + n0 + tx];
    __syncthreads();
    wt[(size_t)(n0 + ty) * HSZ + k0 + tx] = (__bf16)tile[tx][ty];
}

// ------------------------------------------------------- QKV GEMM: C = A*Bt + bias
// 128x128 tile per workgroup; A/B panels double-buffered in LDS via async-to-LDS;
// each wave32 computes a 32x64 slab (8 accumulators, v_wmma_f32_16x16x32_bf16).
__global__ __launch_bounds__(256) void k_gemm_qkv(const __bf16* __restrict__ A,
                                                  const __bf16* __restrict__ Bt,
                                                  const float*  __restrict__ bias,
                                                  float* __restrict__ C)
{
    __shared__ __align__(16) __bf16 sA[2][128][LDSW];
    __shared__ __align__(16) __bf16 sB[2][128][LDSW];

    const int tid   = threadIdx.x;
    const int lane  = tid & 31;
    const int wid   = tid >> 5;          // 0..7
    const int lrow  = lane & 15;
    const int lhalf = lane >> 4;
    const int mb    = blockIdx.x;        // 17 blocks of 128 rows
    const int nb    = blockIdx.y;        // 24 blocks of 128 cols

    const __bf16* Ap = A  + (size_t)(mb * 128) * HSZ;   // panel base
    const __bf16* Bp = Bt + (size_t)(nb * 128) * HSZ;

    // each thread stages 2x16B of A and 2x16B of B per k-step
    const int srow  = tid >> 1;          // 0..127
    const int cpair = (tid & 1) * 2;     // chunk 0 or 2 (16B chunks)

    auto issue = [&](int buf, int k) {
        const char* ga = (const char*)(Ap + (size_t)srow * HSZ + k) + cpair * 16;
        const char* gb = (const char*)(Bp + (size_t)srow * HSZ + k) + cpair * 16;
        uint32_t la = (uint32_t)(uintptr_t)&sA[buf][srow][cpair * 8];
        uint32_t lb = (uint32_t)(uintptr_t)&sB[buf][srow][cpair * 8];
        async_b128(la,      ga);
        async_b128(la + 16, ga + 16);
        async_b128(lb,      gb);
        async_b128(lb + 16, gb + 16);
    };

    const int wm = wid & 3;              // m-group: rows wm*32 .. +31
    const int wn = wid >> 2;             // n-group: cols wn*64 .. +63
    v8f acc[2][4] = {};

    issue(0, 0);
    const int NK = HSZ / 32;             // 64 k-steps
    for (int kt = 0; kt < NK; ++kt) {
        wait_async0();
        __syncthreads();
        if (kt + 1 < NK) issue((kt + 1) & 1, (kt + 1) * 32);

        const int b = kt & 1;
        v16bf af[2];
#pragma unroll
        for (int i = 0; i < 2; ++i) {
            const int r = wm * 32 + i * 16 + lrow;
            v8bf lo = *(const v8bf*)(&sA[b][r][lhalf * 8]);
            v8bf hi = *(const v8bf*)(&sA[b][r][lhalf * 8 + 16]);
            af[i] = __builtin_shufflevector(lo, hi, 0,1,2,3,4,5,6,7,8,9,10,11,12,13,14,15);
        }
#pragma unroll
        for (int j = 0; j < 4; ++j) {
            const int r = wn * 64 + j * 16 + lrow;
            v8bf lo = *(const v8bf*)(&sB[b][r][lhalf * 8]);
            v8bf hi = *(const v8bf*)(&sB[b][r][lhalf * 8 + 16]);
            v16bf bf = __builtin_shufflevector(lo, hi, 0,1,2,3,4,5,6,7,8,9,10,11,12,13,14,15);
#pragma unroll
            for (int i = 0; i < 2; ++i)
                acc[i][j] = __builtin_amdgcn_wmma_f32_16x16x32_bf16(false, af[i], false, bf,
                                                                    (short)0, acc[i][j], false, false);
        }
    }

    // epilogue: C layout comp jc -> row tile_m + jc + 8*lhalf, col tile_n + lrow
#pragma unroll
    for (int i = 0; i < 2; ++i) {
        const int rbase = mb * 128 + wm * 32 + i * 16 + (lhalf << 3);
#pragma unroll
        for (int j = 0; j < 4; ++j) {
            const int col = nb * 128 + wn * 64 + j * 16 + lrow;
            const float bv = bias[col];
            float* crow = C + (size_t)rbase * NOUT + col;
#pragma unroll
            for (int jc = 0; jc < 8; ++jc)
                crow[(size_t)jc * NOUT] = acc[i][j][jc] + bv;
        }
    }
}

// --------------------------------------------- RoPE + pack Qb/Kb [h][t][d], Vt [h][d][t]
__global__ void k_rope_pack(const float* __restrict__ qkv, const int* __restrict__ pos,
                            __bf16* __restrict__ Qb, __bf16* __restrict__ Kb,
                            __bf16* __restrict__ Vt)
{
    const int t = blockIdx.x;
    const int r = blockIdx.y;        // 0..15 q-heads, 16..19 k-heads, 20..23 v-heads
    const int d = threadIdx.x;       // 0..63
    const float* row = qkv + (size_t)t * NOUT + r * HD;

    if (r < NH + NKV) {
        const float ang = (float)pos[t] * __expf((float)d * (-9.210340371976184f / 64.0f)); // ln(1e4)
        float sn, cs; __sincosf(ang, &sn, &cs);
        const float x1 = row[d], x2 = row[d + 64];
        const float o1 = x1 * cs - x2 * sn;
        const float o2 = x2 * cs + x1 * sn;
        if (r < NH) {
            __bf16* q = Qb + ((size_t)r * T_TOK + t) * HD;
            q[d]      = (__bf16)(o1 * QK_SCALE);
            q[d + 64] = (__bf16)(o2 * QK_SCALE);
        } else {
            __bf16* k = Kb + ((size_t)(r - NH) * T_TOK + t) * HD;
            k[d]      = (__bf16)o1;
            k[d + 64] = (__bf16)o2;
        }
    } else {
        const int vh = r - NH - NKV;
        Vt[((size_t)vh * HD + d     ) * TPAD + t] = (__bf16)row[d];
        Vt[((size_t)vh * HD + d + 64) * TPAD + t] = (__bf16)row[d + 64];
    }
}

// zero the padded token columns of Vt so padded-P (==0) WMMAs stay finite
__global__ void k_vt_pad(__bf16* __restrict__ Vt)
{
    int i = blockIdx.x * blockDim.x + threadIdx.x;
    if (i >= NKV * HD * (TPAD - T_TOK)) return;
    int tt = i & 31;
    int dh = i >> 5;
    Vt[(size_t)dh * TPAD + T_TOK + tt] = (__bf16)0.0f;
}

// ----------------------------------------------------- attention: one WG per (qtile, head)
__global__ __launch_bounds__(128) void k_attn(const __bf16* __restrict__ Qb,
                                              const __bf16* __restrict__ Kb,
                                              const __bf16* __restrict__ Vt,
                                              float* __restrict__ out)
{
    __shared__ float  sc[16][MAXC];    // raw scores (f32)
    __shared__ __bf16 pb[16][PCOLS];   // normalized probs (bf16), zero-padded to kpad
    __shared__ float  red[16][8];      // softmax partial reductions
    __shared__ float  rowstat[16];     // row max, then row 1/sum

    const int t0    = blockIdx.x * 16;
    const int h     = blockIdx.y;
    const int kvh   = h >> 2;          // GQA: 4 q-heads per kv-head
    const int lane  = threadIdx.x & 31;
    const int wid   = threadIdx.x >> 5;
    const int lrow  = lane & 15;
    const int lhalf = lane >> 4;

    const bool meta = (t0 < MMETA);
    int klo = 0, ncols;
    if (meta) {
        ncols = t0 + 16;                               // causal meta keys only
    } else {
        int a = t0 - (WWIN - 1);
        klo   = (a < MMETA) ? MMETA : (a & ~15);       // first local key token (tile aligned)
        ncols = MMETA + (t0 + 16 - klo);               // 128 meta cols + window cols
    }
    const int ntiles = ncols >> 4;
    const int kpad   = (ncols + 31) & ~31;

    // Q fragments for this wave: 4 steps over head dim (K = 128)
    v16bf qf[4];
    {
        const __bf16* qp = Qb + ((size_t)h * T_TOK + (t0 + lrow)) * HD + (lhalf << 3);
#pragma unroll
        for (int ks = 0; ks < 4; ++ks) {
            v8bf lo = *(const v8bf*)(qp + ks * 32);
            v8bf hi = *(const v8bf*)(qp + ks * 32 + 16);
            qf[ks]  = __builtin_shufflevector(lo, hi, 0,1,2,3,4,5,6,7,8,9,10,11,12,13,14,15);
        }
    }

    // ---- QK^T: waves split key tiles round-robin ----
    for (int kt = wid; kt < ntiles; kt += 4) {
        const int col0 = kt << 4;
        const int tb   = meta ? col0 : (col0 < MMETA ? col0 : klo + col0 - MMETA);
        const __bf16* kp = Kb + ((size_t)kvh * T_TOK + (tb + lrow)) * HD + (lhalf << 3);
        v8f acc = {};
#pragma unroll
        for (int ks = 0; ks < 4; ++ks) {
            v8bf blo = *(const v8bf*)(kp + ks * 32);
            v8bf bhi = *(const v8bf*)(kp + ks * 32 + 16);
            v16bf b = __builtin_shufflevector(blo, bhi, 0,1,2,3,4,5,6,7,8,9,10,11,12,13,14,15);
            acc = __builtin_amdgcn_wmma_f32_16x16x32_bf16(false, qf[ks], false, b,
                                                          (short)0, acc, false, false);
        }
#pragma unroll
        for (int j = 0; j < 8; ++j)
            sc[j + (lhalf << 3)][col0 + lrow] = acc[j];
    }
    __syncthreads();

    // ---- masked softmax: 8 threads per query row ----
    {
        const int srow = threadIdx.x >> 3;     // 0..15
        const int slot = threadIdx.x & 7;      // 0..7
        const int t = t0 + srow;

        float pm = -3.0e38f;
        for (int c = slot; c < ncols; c += 8) {
            const int s = meta ? c : (c < MMETA ? c : klo + c - MMETA);
            bool valid;
            if (meta)            valid = (s <= t);
            else if (c < MMETA)  valid = true;                         // meta keys unmasked
            else                 valid = (s <= t) && (t - s < WWIN);   // sliding window
            const float v = valid ? sc[srow][c] : -1.0e30f;
            sc[srow][c] = v;
            pm = fmaxf(pm, v);
        }
        red[srow][slot] = pm;
        __syncthreads();
        if (slot == 0) {
            float m = red[srow][0];
#pragma unroll
            for (int i = 1; i < 8; ++i) m = fmaxf(m, red[srow][i]);
            rowstat[srow] = m;
        }
        __syncthreads();
        const float mx = rowstat[srow];
        float ps = 0.0f;
        for (int c = slot; c < ncols; c += 8) {
            const float e = __expf(sc[srow][c] - mx);
            sc[srow][c] = e;
            ps += e;
        }
        red[srow][slot] = ps;
        __syncthreads();
        if (slot == 0) {
            float s = red[srow][0];
#pragma unroll
            for (int i = 1; i < 8; ++i) s += red[srow][i];
            rowstat[srow] = 1.0f / s;
        }
        __syncthreads();
        const float inv = rowstat[srow];
        for (int c = slot; c < ncols; c += 8)
            pb[srow][c] = (__bf16)(sc[srow][c] * inv);
        for (int c = ncols + slot; c < kpad; c += 8)
            pb[srow][c] = (__bf16)0.0f;
    }
    __syncthreads();

    // ---- P @ V: each wave owns 2 d-tiles (16x32 slice of the 16x128 output) ----
    v8f oacc[2] = { {}, {} };
    const int dt0 = wid * 2;
    for (int c0 = 0; c0 < kpad; c0 += 32) {
        const int tb = meta ? c0 : (c0 < MMETA ? c0 : klo + c0 - MMETA);
        v8bf plo = *(const v8bf*)(&pb[lrow][c0 + (lhalf << 3)]);
        v8bf phi = *(const v8bf*)(&pb[lrow][c0 + (lhalf << 3) + 16]);
        v16bf pa = __builtin_shufflevector(plo, phi, 0,1,2,3,4,5,6,7,8,9,10,11,12,13,14,15);
#pragma unroll
        for (int i = 0; i < 2; ++i) {
            const int d = (dt0 + i) * 16 + lrow;
            const __bf16* vp = Vt + ((size_t)kvh * HD + d) * TPAD + tb + (lhalf << 3);
            v8bf vlo = *(const v8bf*)(vp);
            v8bf vhi = *(const v8bf*)(vp + 16);
            v16bf vb = __builtin_shufflevector(vlo, vhi, 0,1,2,3,4,5,6,7,8,9,10,11,12,13,14,15);
            oacc[i] = __builtin_amdgcn_wmma_f32_16x16x32_bf16(false, pa, false, vb,
                                                              (short)0, oacc[i], false, false);
        }
    }
    float* orow = out + (size_t)(t0 + (lhalf << 3)) * (NH * HD) + h * HD;
#pragma unroll
    for (int i = 0; i < 2; ++i) {
        const int d = (dt0 + i) * 16 + lrow;
#pragma unroll
        for (int j = 0; j < 8; ++j)
            orow[(size_t)j * (NH * HD) + d] = oacc[i][j];
    }
}

// ------------------------------------------------------------------ launch
extern "C" void kernel_launch(void* const* d_in, const int* in_sizes, int n_in,
                              void* d_out, int out_size, void* d_ws, size_t ws_size,
                              hipStream_t stream)
{
    const float* hidden = (const float*)d_in[0];
    const float* Wqkv   = (const float*)d_in[1];
    const float* bqkv   = (const float*)d_in[2];
    const int*   pos    = (const int*)d_in[3];
    float* out = (float*)d_out;

    char* p = (char*)d_ws;
    auto alloc = [&](size_t bytes) {
        char* r = p;
        p += (bytes + 255) & ~(size_t)255;
        return r;
    };
    __bf16* hb  = (__bf16*)alloc((size_t)T_TOK * HSZ * 2);
    __bf16* wt  = (__bf16*)alloc((size_t)NOUT * HSZ * 2);
    float*  qkv = (float*) alloc((size_t)T_TOK * NOUT * 4);
    __bf16* Qb  = (__bf16*)alloc((size_t)NH  * T_TOK * HD * 2);
    __bf16* Kb  = (__bf16*)alloc((size_t)NKV * T_TOK * HD * 2);
    __bf16* Vt  = (__bf16*)alloc((size_t)NKV * HD * TPAD * 2);

    {   // hidden -> bf16
        int n4 = T_TOK * HSZ / 4;
        k_cvt_hidden<<<(n4 + 255) / 256, 256, 0, stream>>>(hidden, hb, n4);
    }
    {   // W -> Wt bf16
        dim3 g(NOUT / 32, HSZ / 32), b(32, 32);
        k_transpose_w<<<g, b, 0, stream>>>(Wqkv, wt);
    }
    {   // qkv = hb @ wt^T + bias, 128x128 tiles, async-to-LDS double buffered
        dim3 g(T_TOK / 128, NOUT / 128);
        k_gemm_qkv<<<g, 256, 0, stream>>>(hb, wt, bqkv, qkv);
    }
    {   // RoPE + pack
        dim3 g(T_TOK, NH + 2 * NKV);
        k_rope_pack<<<g, 64, 0, stream>>>(qkv, pos, Qb, Kb, Vt);
    }
    {   // zero Vt token padding
        int n = NKV * HD * (TPAD - T_TOK);
        k_vt_pad<<<(n + 255) / 256, 256, 0, stream>>>(Vt);
    }
    {   // attention
        dim3 g(T_TOK / 16, NH);
        k_attn<<<g, 128, 0, stream>>>(Qb, Kb, Vt, out);
    }
}
